// TemporalPromptGeneratorModel_18966575579194
// MI455X (gfx1250) — matrix operand order repaired and checked by reading, not compiled, not run on hardware
//
#include <hip/hip_runtime.h>
#include <hip/hip_bf16.h>

// MI455X (gfx1250) implementation.
// Dead code eliminated: softmax over singleton axis -> attn path folds to one
// 400x400 matrix; h0==0 -> GRU recurrent matrices unused. Remaining ~925 GFLOP
// of GEMM work runs through v_wmma_f32_16x16x32_f16 (f16 inputs, f32 accum),
// the fast dense matrix path on CDNA5 wave32.

typedef __attribute__((ext_vector_type(16))) _Float16 v16h;
typedef __attribute__((ext_vector_type(8)))  _Float16 v8h;
typedef __attribute__((ext_vector_type(8)))  float    v8f;

#define LDSK 40   // 32 k-halfs + 8 pad (80B row stride, 16B aligned, spreads banks)

// ---------------------------------------------------------------------------
// Fold Wv(400x6400) @ Wo(6400x400) -> Meff(400x400)
__global__ __launch_bounds__(256) void meff_kernel(
    const float* __restrict__ Wv, const float* __restrict__ Wo,
    float* __restrict__ Meff) {
  int i = blockIdx.x >> 1;
  int d = ((blockIdx.x & 1) << 8) + threadIdx.x;
  if (d >= 400) return;
  const float* wv = Wv + (long)i * 6400;
  float acc = 0.f;
  for (int j = 0; j < 6400; j++) acc += wv[j] * Wo[(long)j * 400 + d];
  Meff[(long)i * 400 + d] = acc;
}

__global__ __launch_bounds__(256) void beff_kernel(
    const float* __restrict__ bv, const float* __restrict__ Wo,
    const float* __restrict__ bo, float* __restrict__ beff) {
  int d = blockIdx.x * 256 + threadIdx.x;
  if (d >= 400) return;
  float acc = bo[d];
  for (int j = 0; j < 6400; j++) acc += bv[j] * Wo[(long)j * 400 + d];
  beff[d] = acc;
}

// ---------------------------------------------------------------------------
__device__ inline void blockReduce2(float* r1, float* r2, int tid,
                                    float a, float b, float& s1, float& s2) {
  __syncthreads();
  r1[tid] = a; r2[tid] = b;
  __syncthreads();
  for (int s = 256; s > 0; s >>= 1) {
    if (tid < s) { r1[tid] += r1[tid + s]; r2[tid] += r2[tid + s]; }
    __syncthreads();
  }
  s1 = r1[0]; s2 = r2[0];
}

// Encoder: attn (folded) + LN1 + rank-2 FFN + LN2; builds f16 flat buffer
// flat[b] = [x[b,0,:], x[b,1,:], x[b,2,:], output[b]] (1600 halfs)
__global__ __launch_bounds__(512) void encoder_kernel(
    const float* __restrict__ x,
    const float* __restrict__ Meff, const float* __restrict__ beff,
    const float* __restrict__ g1, const float* __restrict__ b1,
    const float* __restrict__ w1, const float* __restrict__ fb1,
    const float* __restrict__ w2, const float* __restrict__ fb2,
    const float* __restrict__ g2, const float* __restrict__ b2,
    _Float16* __restrict__ flat16) {
  __shared__ float xs[400];
  __shared__ float r1[512];
  __shared__ float r2[512];
  const int tid = threadIdx.x;
  const long b = blockIdx.x;
  const float* xb = x + b * 1200;
  _Float16* fl = flat16 + b * 1600;
  for (int i = tid; i < 1200; i += 512) fl[i] = (_Float16)xb[i];
  if (b == 0) {  // output[0] = x[0, 2, :]
    for (int i = tid; i < 400; i += 512) fl[1200 + i] = (_Float16)xb[800 + i];
    return;
  }
  if (tid < 400) xs[tid] = xb[800 + tid];
  __syncthreads();
  float attn = 0.f;
  if (tid < 400) {
    attn = beff[tid];
    for (int i = 0; i < 400; i++) attn += xs[i] * Meff[i * 400 + tid];
  }
  float h = (tid < 400) ? xs[tid] + attn : 0.f;
  float s1, s2;
  blockReduce2(r1, r2, tid, h, h * h, s1, s2);
  float mean = s1 * (1.f / 400.f);
  float var  = s2 * (1.f / 400.f) - mean * mean;
  float rstd = rsqrtf(var + 1e-6f);
  float o = (tid < 400) ? (h - mean) * rstd * g1[tid] + b1[tid] : 0.f;
  // rank-2 FFN: two dot products
  blockReduce2(r1, r2, tid,
               (tid < 400) ? o * w1[tid * 2 + 0] : 0.f,
               (tid < 400) ? o * w1[tid * 2 + 1] : 0.f, s1, s2);
  float t0 = s1 + fb1[0]; t0 = t0 > 0.f ? t0 : 0.f;
  float t1 = s2 + fb1[1]; t1 = t1 > 0.f ? t1 : 0.f;
  float f = (tid < 400) ? o + t0 * w2[tid] + t1 * w2[400 + tid] + fb2[tid] : 0.f;
  blockReduce2(r1, r2, tid, f, f * f, s1, s2);
  mean = s1 * (1.f / 400.f);
  var  = s2 * (1.f / 400.f) - mean * mean;
  rstd = rsqrtf(var + 1e-6f);
  if (tid < 400) fl[1200 + tid] = (_Float16)((f - mean) * rstd * g2[tid] + b2[tid]);
}

// ---------------------------------------------------------------------------
// Transpose + f32->f16 convert: src (K x N, row-major f32) -> dst (Np x K f16)
// dst[n][k] = n<N ? src[k][n] : 0  (zero-pad rows so GEMMs run dense)
__global__ __launch_bounds__(256) void transpose_cvt(
    const float* __restrict__ src, int K, int N,
    _Float16* __restrict__ dst, int Np) {
  __shared__ float t[32][33];
  const int tx = threadIdx.x, ty = threadIdx.y;
  const int kb = blockIdx.x * 32, nb = blockIdx.y * 32;
  #pragma unroll
  for (int r = 0; r < 4; r++) {
    int k = kb + ty + r * 8, n = nb + tx;
    t[ty + r * 8][tx] = (k < K && n < N) ? src[(long)k * N + n] : 0.f;
  }
  __syncthreads();
  #pragma unroll
  for (int r = 0; r < 4; r++) {
    int n = nb + ty + r * 8, k = kb + tx;
    if (n < Np && k < K) dst[(long)n * K + k] = (_Float16)t[tx][ty + r * 8];
  }
}

// ---------------------------------------------------------------------------
// Generic WMMA GEMM: C(MxN) = A(MxK) * B^T where B is stored N-major (Np x K).
// Block tile 128x128, 8 waves (2 M x 4 N), each wave 64x32 = 8 wmma tiles.
// LDS double-buffered, K remainder zero-filled (K must be a multiple of 8).
// mode 0: store f16 raw; mode 1: f16 elu(acc+bias[n]); mode 2: f32 acc+bias[n]
__global__ __launch_bounds__(256) void gemm_wmma_f16(
    const _Float16* __restrict__ A, long lda,
    const _Float16* __restrict__ B, long ldb,
    void* __restrict__ C, long ldc, long coff,
    const float* __restrict__ bias,
    int K, int Nreal, int mode) {
  __shared__ __align__(16) _Float16 As[2][128 * LDSK];
  __shared__ __align__(16) _Float16 Bs[2][128 * LDSK];
  const int tid  = threadIdx.x;
  const int lane = tid & 31;
  const int wave = tid >> 5;
  const int wm = wave & 1;    // 0..1 -> 64-row slice
  const int wn = wave >> 1;   // 0..3 -> 32-col slice
  const long m0 = (long)blockIdx.x * 128;
  const long n0 = (long)blockIdx.y * 128;
  const int l16 = lane & 15, lh = lane >> 4;

  v8f acc[4][2];
  v8f zf = {0.f, 0.f, 0.f, 0.f, 0.f, 0.f, 0.f, 0.f};
  #pragma unroll
  for (int i = 0; i < 4; i++)
    #pragma unroll
    for (int j = 0; j < 2; j++) acc[i][j] = zf;

  const int nk = (K + 31) >> 5;
  v8h zero8 = {};
  v8h pa[2], pb[2];

  auto loadStage = [&](int kt) {
    #pragma unroll
    for (int i = 0; i < 2; i++) {
      int u = tid + 256 * i;
      int row = u >> 2, seg = u & 3;
      int kk = kt * 32 + seg * 8;
      pa[i] = (kk < K) ? *(const v8h*)(A + (m0 + row) * lda + kk) : zero8;
      pb[i] = (kk < K) ? *(const v8h*)(B + (n0 + row) * ldb + kk) : zero8;
    }
  };
  auto storeStage = [&](int bi) {
    #pragma unroll
    for (int i = 0; i < 2; i++) {
      int u = tid + 256 * i;
      int row = u >> 2, seg = u & 3;
      *(v8h*)(As[bi] + row * LDSK + seg * 8) = pa[i];
      *(v8h*)(Bs[bi] + row * LDSK + seg * 8) = pb[i];
    }
  };
  auto compute = [&](int bi) {
    v16h af[4], bf[2];
    #pragma unroll
    for (int mt = 0; mt < 4; mt++) {
      // A 16x32 f16 layout: lanes 0-15 K[0..7]+[16..23], lanes 16-31 K[8..15]+[24..31]
      const _Float16* p = As[bi] + (wm * 64 + mt * 16 + l16) * LDSK + lh * 8;
      v8h lo = *(const v8h*)p;
      v8h hi = *(const v8h*)(p + 16);
      af[mt] = __builtin_shufflevector(lo, hi, 0,1,2,3,4,5,6,7,8,9,10,11,12,13,14,15);
    }
    #pragma unroll
    for (int nt = 0; nt < 2; nt++) {
      // B 32x16: lane col = lane%16, lanes 0-15 K[0..7]+[16..23], lanes 16-31 K[8..15]+[24..31]
      const _Float16* p = Bs[bi] + (wn * 32 + nt * 16 + l16) * LDSK + lh * 8;
      v8h lo = *(const v8h*)p;
      v8h hi = *(const v8h*)(p + 16);
      bf[nt] = __builtin_shufflevector(lo, hi, 0,1,2,3,4,5,6,7,8,9,10,11,12,13,14,15);
    }
    #pragma unroll
    for (int mt = 0; mt < 4; mt++)
      #pragma unroll
      for (int nt = 0; nt < 2; nt++)
        acc[mt][nt] = __builtin_amdgcn_wmma_f32_16x16x32_f16(
            false, af[mt], false, bf[nt], (short)0, acc[mt][nt], false, false);
  };

  loadStage(0);
  storeStage(0);
  __syncthreads();
  for (int kt = 0; kt < nk; kt++) {
    int cur = kt & 1, nxt = cur ^ 1;
    if (kt + 1 < nk) loadStage(kt + 1);   // global prefetch overlaps wmma
    compute(cur);
    __syncthreads();
    if (kt + 1 < nk) storeStage(nxt);
    __syncthreads();
  }

  // Epilogue. C/D layout: lane holds col n = lane%16; vgpr i -> row i + 8*(lane/16)
  #pragma unroll
  for (int mt = 0; mt < 4; mt++) {
    #pragma unroll
    for (int nt = 0; nt < 2; nt++) {
      long n = n0 + wn * 32 + nt * 16 + l16;
      if (n >= Nreal) continue;
      long rowBase = m0 + wm * 64 + mt * 16 + lh * 8;
      if (mode == 0) {
        _Float16* Cp = (_Float16*)C;
        #pragma unroll
        for (int i = 0; i < 8; i++)
          Cp[(rowBase + i) * ldc + coff + n] = (_Float16)acc[mt][nt][i];
      } else if (mode == 1) {
        float bs = bias[n];
        _Float16* Cp = (_Float16*)C;
        #pragma unroll
        for (int i = 0; i < 8; i++) {
          float v = acc[mt][nt][i] + bs;
          v = v > 0.f ? v : expm1f(v);
          Cp[(rowBase + i) * ldc + coff + n] = (_Float16)v;
        }
      } else {
        float bs = bias[n];
        float* Cp = (float*)C;
        #pragma unroll
        for (int i = 0; i < 8; i++)
          Cp[(rowBase + i) * ldc + coff + n] = acc[mt][nt][i] + bs;
      }
    }
  }
}

// ---------------------------------------------------------------------------
// GRU gate math (h0==0): hf = (1-sigmoid(xz+b1z)) * elu(xh + sigmoid(xr+b1r)*b1h)
// Y1 cols [0,5400)=fwd, [5400,10800)=bwd. Writes x1 into cat[:, 0:3600] f16.
__global__ __launch_bounds__(256) void gru_gates_kernel(
    const _Float16* __restrict__ Y1,
    const float* __restrict__ fb, const float* __restrict__ bb,
    _Float16* __restrict__ cat) {
  long idx = (long)blockIdx.x * 256 + threadIdx.x;
  if (idx >= (long)16384 * 3600) return;
  long b = idx / 3600;
  int col = (int)(idx - b * 3600);
  const float* Bp; int off, u;
  if (col < 1800) { u = col;        off = 0;    Bp = fb; }
  else            { u = col - 1800; off = 5400; Bp = bb; }
  const _Float16* y = Y1 + b * 10880 + off;
  float xz = (float)y[u]        + Bp[u]        + Bp[5400 + u];
  float xr = (float)y[1800 + u] + Bp[1800 + u] + Bp[5400 + 1800 + u];
  float xh = (float)y[3600 + u] + Bp[3600 + u];
  float z = 1.f / (1.f + expf(-xz));
  float r = 1.f / (1.f + expf(-xr));
  float hin = xh + r * Bp[5400 + 3600 + u];
  float hh = hin > 0.f ? hin : expm1f(hin);
  cat[b * 7200 + col] = (_Float16)((1.f - z) * hh);
}

// ---------------------------------------------------------------------------
extern "C" void kernel_launch(void* const* d_in, const int* in_sizes, int n_in,
                              void* d_out, int out_size, void* d_ws, size_t ws_size,
                              hipStream_t stream) {
  (void)in_sizes; (void)n_in; (void)out_size; (void)ws_size;
  // setup_inputs() dict order
  const float* x      = (const float*)d_in[0];
  const float* Wv     = (const float*)d_in[5];
  const float* bv     = (const float*)d_in[6];
  const float* Wo     = (const float*)d_in[7];
  const float* bo     = (const float*)d_in[8];
  const float* ln1_b  = (const float*)d_in[9];
  const float* ffn_w1 = (const float*)d_in[10];
  const float* ffn_b1 = (const float*)d_in[11];
  const float* ffn_w2 = (const float*)d_in[12];
  const float* ffn_b2 = (const float*)d_in[13];
  const float* ln2_b  = (const float*)d_in[14];
  const float* gru_fk = (const float*)d_in[15];
  const float* gru_fb = (const float*)d_in[17];
  const float* gru_bk = (const float*)d_in[18];
  const float* gru_bb = (const float*)d_in[20];
  const float* d2_w   = (const float*)d_in[21];
  const float* d2_b   = (const float*)d_in[22];
  const float* d4_w   = (const float*)d_in[23];
  const float* d4_b   = (const float*)d_in[24];
  const float* out_w  = (const float*)d_in[25];
  const float* out_b  = (const float*)d_in[26];
  const float* ln1_g  = (const float*)d_in[27];
  const float* ln2_g  = (const float*)d_in[28];

  // Workspace carve-up (~705 MB)
  size_t off = 0;
  auto take = [&](size_t bytes) {
    void* p = (char*)d_ws + off;
    off += (bytes + 255) & ~(size_t)255;
    return p;
  };
  float*     Meff   = (float*)take((size_t)400 * 400 * 4);
  float*     beff   = (float*)take((size_t)400 * 4);
  _Float16*  flat16 = (_Float16*)take((size_t)16384 * 1600 * 2);
  _Float16*  W1t    = (_Float16*)take((size_t)10880 * 1600 * 2); // [fwd|bwd|pad] x K
  _Float16*  Y1     = (_Float16*)take((size_t)16384 * 10880 * 2);
  _Float16*  cat    = (_Float16*)take((size_t)16384 * 7200 * 2); // [x1|x3|x5]
  _Float16*  W2t    = (_Float16*)take((size_t)1920 * 3600 * 2);
  _Float16*  W3t    = (_Float16*)take((size_t)1920 * 1800 * 2);
  _Float16*  WOt    = (_Float16*)take((size_t)256 * 7200 * 2);

  // 1. Fold attention weights
  meff_kernel<<<800, 256, 0, stream>>>(Wv, Wo, Meff);
  beff_kernel<<<2, 256, 0, stream>>>(bv, Wo, bo, beff);

  // 2. Encoder + build f16 flat buffer (16384 x 1600)
  encoder_kernel<<<16384, 512, 0, stream>>>(
      x, Meff, beff, ln1_g, ln1_b, ffn_w1, ffn_b1, ffn_w2, ffn_b2,
      ln2_g, ln2_b, flat16);

  // 3. Weight transpose-converts to f16 (N-major, zero-padded rows)
  dim3 tb(32, 8);
  transpose_cvt<<<dim3(50, 169),  tb, 0, stream>>>(gru_fk, 1600, 5400, W1t, 5400);
  transpose_cvt<<<dim3(50, 172),  tb, 0, stream>>>(gru_bk, 1600, 5400,
                                                   W1t + (size_t)5400 * 1600, 5480);
  transpose_cvt<<<dim3(113, 60),  tb, 0, stream>>>(d2_w, 3600, 1800, W2t, 1920);
  transpose_cvt<<<dim3(57, 60),   tb, 0, stream>>>(d4_w, 1800, 1800, W3t, 1920);
  transpose_cvt<<<dim3(225, 8),   tb, 0, stream>>>(out_w, 7200, 140, WOt, 256);

  // 4. GEMM1: Y1 = flat @ [gru_fk | gru_bk]   (16384 x 1600 x 10800)
  gemm_wmma_f16<<<dim3(128, 85), 256, 0, stream>>>(
      flat16, 1600, W1t, 1600, Y1, 10880, 0, nullptr, 1600, 10880, 0);

  // 5. GRU gates -> x1 = cat[:, 0:3600]
  gru_gates_kernel<<<230400, 256, 0, stream>>>(Y1, gru_fb, gru_bb, cat);

  // 6. GEMM2: x3 = elu(x1 @ d2_w + d2_b) -> cat[:, 3600:5400]
  gemm_wmma_f16<<<dim3(128, 15), 256, 0, stream>>>(
      cat, 7200, W2t, 3600, cat, 7200, 3600, d2_b, 3600, 1800, 1);

  // 7. GEMM3: x5 = elu(x3 @ d4_w + d4_b) -> cat[:, 5400:7200]
  gemm_wmma_f16<<<dim3(128, 15), 256, 0, stream>>>(
      cat + 3600, 7200, W3t, 1800, cat, 7200, 5400, d4_b, 1800, 1800, 1);

  // 8. GEMM4: out = cat @ out_w + out_b (f32, N=140 predicated)
  gemm_wmma_f16<<<dim3(128, 2), 256, 0, stream>>>(
      cat, 7200, WOt, 7200, d_out, 140, 0, out_b, 7200, 140, 2);
}
